// K_UNIT_54451595379234
// MI455X (gfx1250) — matrix operand order repaired and checked by reading, not compiled, not run on hardware
//
#include <hip/hip_runtime.h>
#include <hip/hip_bf16.h>

typedef __attribute__((ext_vector_type(16))) _Float16 v16h;
typedef __attribute__((ext_vector_type(8)))  float    v8f;

#define HH   256
#define WCC  512      // W*C
#define NCOL 1024     // B*W*C
#define WPB  8        // waves per block

__global__ __launch_bounds__(256) void K_sinc_warp_wmma(
    const float* __restrict__ img, const float* __restrict__ fmap,
    float* __restrict__ out)
{
  __shared__ __align__(16) float    lds_x[WPB][HH];   // clipped source coords
  __shared__ __align__(16) float    lds_s[WPB][HH];   // sin(pi * x)
  __shared__ __align__(16) _Float16 lds_ih[WPB][HH];  // img column in f16 (B operand)

  const int tid  = threadIdx.x;
  const int wave = tid >> 5;
  const int lane = tid & 31;
  const int col  = blockIdx.x * WPB + wave;      // 0..1023
  const int b    = col >> 9;                     // col / (W*C)
  const int wc   = col & 511;
  const long long base = (long long)b * HH * WCC + wc;

  const float PI = 3.14159274101257324f;  // float32(pi)

  // ---- Phase 1: stage column: x = clip(fmap+i, 0, H-1); s = sin(pi*x); img->f16
  #pragma unroll
  for (int r = 0; r < 8; ++r) {
    const int i = lane + 32 * r;
    const float fv = fmap[base + (long long)i * WCC];
    const float iv = img [base + (long long)i * WCC];
    float x = fv + (float)i;
    x = fminf(fmaxf(x, 0.0f), 255.0f);
    lds_x[wave][i]  = x;
    // v_sin_f32 input is radians/(2*pi): sin(pi*x) == hwsin(0.5*x)
    lds_s[wave][i]  = __builtin_amdgcn_sinf(0.5f * x);
    lds_ih[wave][i] = (_Float16)iv;
  }
  __syncthreads();

  const int   m     = lane & 15;     // A-matrix row (j within tile)
  const int   hi    = lane >> 4;     // lane half selects K sub-ranges
  const int   kbase = hi * 8;
  const float fm    = (float)m;
  // fold (-1)^j into the reciprocal's denominator: sgn/(pi*d) == 1/((sgn*pi)*d)
  const float pisgn = (m & 1) ? -PI : PI;

  const unsigned int* ihw = (const unsigned int*)&lds_ih[wave][0];

  // 16 independent accumulators: one per j-tile (128 VGPRs, lives across it-loop)
  v8f acc[16];
  {
    v8f z = {};
    #pragma unroll
    for (int jt = 0; jt < 16; ++jt) acc[jt] = z;
  }

  // ---- Phase 2: outer i-tiles (rolled), inner j-tiles (unrolled, per-jt acc)
  #pragma clang loop unroll(disable)
  for (int it = 0; it < 8; ++it) {
    const int i0 = it * 32;

    // B fragment (img broadcast along N): lanes0-15 k=0..15, lanes16-31 k=16..31.
    // Load ONCE per i-tile; the empty asm makes the dwords opaque so the
    // compiler cannot re-sink the LDS loads next to each WMMA (round-3 bubble).
    unsigned int bu[8];
    {
      const int bb = (i0 + hi * 16) >> 1;   // packed-pair dword index
      #pragma unroll
      for (int p = 0; p < 8; ++p) bu[p] = ihw[bb + p];
      asm volatile("" : "+v"(bu[0]), "+v"(bu[1]), "+v"(bu[2]), "+v"(bu[3]),
                        "+v"(bu[4]), "+v"(bu[5]), "+v"(bu[6]), "+v"(bu[7]));
    }
    v16h bfrag;
    __builtin_memcpy(&bfrag, bu, sizeof(bu));

    // Stage this tile's x/s per lane: k in {16*hi..16*hi+15}
    float xv[16], sv[16];
    #pragma unroll
    for (int t = 0; t < 16; ++t) {
      const int k = hi * 16 + t;
      xv[t] = lds_x[wave][i0 + k];
      sv[t] = lds_s[wave][i0 + k];
    }

    #pragma unroll
    for (int jt = 0; jt < 16; ++jt) {
      const float jf = fm + (float)(jt * 16);
      v16h a;
      #pragma unroll
      for (int t = 0; t < 16; ++t) {
        float d = xv[t] - jf;          // exact near d==0 (Sterbenz), keeps sinc->1
        // d==0: reference value sin(PI)/PI ~ -2.8e-8 underflows f16 to 0;
        // s ~ 0 there anyway; just avoid NaN from 0*rcp(0).
        d = (d == 0.0f) ? 1.0f : d;                 // branchless v_cndmask
        const float r = __builtin_amdgcn_rcpf(pisgn * d);
        a[t] = (_Float16)(sv[t] * r);
      }
      acc[jt] = __builtin_amdgcn_wmma_f32_16x16x32_f16(
          /*neg_a=*/false, a, /*neg_b=*/false, bfrag,
          /*c_mod=*/(short)0, acc[jt], /*reuse_a=*/false, /*reuse_b=*/false);
    }
  }

  // ---- Store: D column N=0 lives in lane 0 (M=0..7) and lane 16 (M=8..15)
  if (m == 0) {
    #pragma unroll
    for (int jt = 0; jt < 16; ++jt) {
      const int j0 = jt * 16 + hi * 8;
      #pragma unroll
      for (int v = 0; v < 8; ++v)
        out[base + (long long)(j0 + v) * WCC] = acc[jt][v];
    }
  }
}

extern "C" void kernel_launch(void* const* d_in, const int* in_sizes, int n_in,
                              void* d_out, int out_size, void* d_ws, size_t ws_size,
                              hipStream_t stream) {
  (void)in_sizes; (void)n_in; (void)out_size; (void)d_ws; (void)ws_size;
  const float* img  = (const float*)d_in[0];
  const float* fmap = (const float*)d_in[1];
  float* out = (float*)d_out;
  dim3 grid(NCOL / WPB);   // 128 blocks
  dim3 block(256);         // 8 wave32 per block
  K_sinc_warp_wmma<<<grid, block, 0, stream>>>(img, fmap, out);
}
